// model_grace_38001870635085
// MI455X (gfx1250) — matrix operand order repaired and checked by reading, not compiled, not run on hardware
//
#include <hip/hip_runtime.h>
#include <hip/hip_bf16.h>
#include <math.h>

// ---------------------------------------------------------------------------
// GRACE contrastive loss, N=16384, H=P=256, tau=0.5.
// Pipeline:
//   1) proj_norm_kernel : fused fc1+elu+fc2+row-normalize, fp32 WMMA
//                         (v_wmma_f32_16x16x4_f32), writes bf16 n1/n2.
//   2) gram_expsum_kernel: streaming exp-row/col-sums of the N x N Gram
//                          matrices via v_wmma_f32_16x16x32_bf16. Each wave
//                          holds TWO 16-row A tile sets in registers so every
//                          B fragment feeds 2 WMMAs (32 FLOP/byte from cache).
//   3) finalize_kernel  : diag dots + log-denoms -> scalar mean loss.
// ---------------------------------------------------------------------------

typedef float v2f  __attribute__((ext_vector_type(2)));
typedef float v8f  __attribute__((ext_vector_type(8)));
typedef __bf16 v16bf __attribute__((ext_vector_type(16)));
typedef unsigned short us8  __attribute__((ext_vector_type(8)));
typedef unsigned short us16 __attribute__((ext_vector_type(16)));

#define NROWS 16384
#define HDIM  256
#define PDIM  256
#define ITAU  2.0f            // 1/tau
#define EXP_INV_TAU 7.38905609893065f  // exp(1/tau)

__device__ __forceinline__ unsigned short f2bf(float f) {
  unsigned int u = __float_as_uint(f);
  u += 0x7fffu + ((u >> 16) & 1u);      // round-to-nearest-even
  return (unsigned short)(u >> 16);
}
__device__ __forceinline__ float bf2f(unsigned short h) {
  return __uint_as_float(((unsigned int)h) << 16);
}

// ---------------------------------------------------------------------------
// Kernel 1: fused projection + normalize.
// Block = 256 threads (8 waves). Block owns 16 full rows; wave w owns cols
// [32w, 32w+32) as two 16-col WMMA tiles. fc1 intermediate (elu output) lives
// in LDS; fc2 output is bias-added, row-norm reduced across the block in LDS,
// then written as bf16 unit vectors.
// ---------------------------------------------------------------------------
__global__ __launch_bounds__(256)
void proj_norm_kernel(const float* __restrict__ z1, const float* __restrict__ z2,
                      const float* __restrict__ w1, const float* __restrict__ b1,
                      const float* __restrict__ w2, const float* __restrict__ b2,
                      unsigned short* __restrict__ n1out,
                      unsigned short* __restrict__ n2out)
{
  const float* z = blockIdx.y ? z2 : z1;
  unsigned short* nout = blockIdx.y ? n2out : n1out;
  const int i0   = blockIdx.x * 16;
  const int tid  = threadIdx.x;
  const int wid  = tid >> 5;
  const int lane = tid & 31;
  const int nn   = lane & 15;   // A-row index / B-col index within a tile
  const int half = lane >> 4;

  __shared__ float u_lds[16][264];   // elu(fc1) tile, padded stride vs banks
  __shared__ float norm_lds[16];

  if (tid < 16) norm_lds[tid] = 0.0f;

  const int col0 = wid * 32;

  // ---- stage 1: u = elu(z @ w1^T + b1) --------------------------------
  v8f c0 = {}; v8f c1 = {};
  const float* zrow = z  + (size_t)(i0 + nn)        * HDIM;
  const float* w1r0 = w1 + (size_t)(col0 + nn)      * HDIM;
  const float* w1r1 = w1 + (size_t)(col0 + 16 + nn) * HDIM;
  #pragma unroll 8
  for (int k = 0; k < HDIM; k += 4) {
    v2f a  = *(const v2f*)(zrow + k + 2 * half);
    v2f bA = *(const v2f*)(w1r0 + k + 2 * half);
    v2f bB = *(const v2f*)(w1r1 + k + 2 * half);
    c0 = __builtin_amdgcn_wmma_f32_16x16x4_f32(false, a, false, bA, (short)0, c0, false, false);
    c1 = __builtin_amdgcn_wmma_f32_16x16x4_f32(false, a, false, bB, (short)0, c1, false, false);
  }
  {
    float bias0 = b1[col0 + nn];
    float bias1 = b1[col0 + 16 + nn];
    #pragma unroll
    for (int r = 0; r < 8; ++r) {
      int m = r + 8 * half;            // C/D layout: VGPR r, half -> row
      float v0 = c0[r] + bias0;
      float v1 = c1[r] + bias1;
      v0 = v0 > 0.0f ? v0 : expm1f(v0);
      v1 = v1 > 0.0f ? v1 : expm1f(v1);
      u_lds[m][col0 + nn]      = v0;
      u_lds[m][col0 + 16 + nn] = v1;
    }
  }
  __syncthreads();

  // ---- stage 2: h = u @ w2^T + b2 -------------------------------------
  v8f h0 = {}; v8f h1 = {};
  const float* w2r0 = w2 + (size_t)(col0 + nn)      * PDIM;
  const float* w2r1 = w2 + (size_t)(col0 + 16 + nn) * PDIM;
  #pragma unroll 8
  for (int k = 0; k < PDIM; k += 4) {
    v2f a  = *(const v2f*)&u_lds[nn][k + 2 * half];
    v2f bA = *(const v2f*)(w2r0 + k + 2 * half);
    v2f bB = *(const v2f*)(w2r1 + k + 2 * half);
    h0 = __builtin_amdgcn_wmma_f32_16x16x4_f32(false, a, false, bA, (short)0, h0, false, false);
    h1 = __builtin_amdgcn_wmma_f32_16x16x4_f32(false, a, false, bB, (short)0, h1, false, false);
  }

  float hb0 = b2[col0 + nn];
  float hb1 = b2[col0 + 16 + nn];
  float hv0[8], hv1[8];
  #pragma unroll
  for (int r = 0; r < 8; ++r) {
    hv0[r] = h0[r] + hb0;
    hv1[r] = h1[r] + hb1;
  }
  // row-wise sum of squares -> LDS (rows r+8*half; reduce over 16 lanes)
  #pragma unroll
  for (int r = 0; r < 8; ++r) {
    float s = hv0[r] * hv0[r] + hv1[r] * hv1[r];
    s += __shfl_xor(s, 1, 32);
    s += __shfl_xor(s, 2, 32);
    s += __shfl_xor(s, 4, 32);
    s += __shfl_xor(s, 8, 32);
    if (nn == 0) atomicAdd(&norm_lds[r + 8 * half], s);
  }
  __syncthreads();

  #pragma unroll
  for (int r = 0; r < 8; ++r) {
    int m = r + 8 * half;
    float inv = 1.0f / fmaxf(sqrtf(norm_lds[m]), 1e-12f);
    size_t base = (size_t)(i0 + m) * PDIM;
    nout[base + col0 + nn]      = f2bf(hv0[r] * inv);
    nout[base + col0 + 16 + nn] = f2bf(hv1[r] * inv);
  }
}

// ---------------------------------------------------------------------------
// Kernel 2: exp row/col sums of exp((nA @ nB^T)/tau).
// Block = 256 threads (8 waves), block tile 256 rows x 128 cols.
// Wave w owns rows [32w, 32w+32): TWO 16-row A tile sets preloaded in
// registers (2 x 64 VGPRs), so each B fragment is consumed by 2 WMMAs.
// K=256 in 8 bf16 WMMA steps. Row sums reduce in registers then 4 shfl_xor;
// column sums (only needed for S12) reduce via shfl_xor(16) + ds_add_f32 +
// one global atomic per column per block.
// ---------------------------------------------------------------------------
__global__ __launch_bounds__(256)
void gram_expsum_kernel(const unsigned short* __restrict__ nA,
                        const unsigned short* __restrict__ nB,
                        float* __restrict__ rowsum,
                        float* __restrict__ colsum,
                        int wantCol)
{
  const int tid  = threadIdx.x;
  const int wid  = tid >> 5;
  const int lane = tid & 31;
  const int nn   = lane & 15;
  const int half = lane >> 4;
  const int rowBase = blockIdx.y * 256 + wid * 32;   // 2 row tiles per wave
  const int colBase = blockIdx.x * 128;

  __shared__ float col_lds[128];
  if (tid < 128) col_lds[tid] = 0.0f;
  __syncthreads();

  // Preload A rows (rowBase+nn, rowBase+16+nn) for all 8 K-steps.
  // bf16 A 16x32 layout: elems 0..7 -> K = 8*half..+7 ; elems 8..15 -> K = 16+8*half..+7
  v16bf afrag[2][8];
  #pragma unroll
  for (int t = 0; t < 2; ++t) {
    const us8* arow = (const us8*)(nA + (size_t)(rowBase + t * 16 + nn) * PDIM);
    #pragma unroll
    for (int kt = 0; kt < 8; ++kt) {
      us8 lo = arow[kt * 4 + half];
      us8 hi = arow[kt * 4 + 2 + half];
      us16 v = __builtin_shufflevector(lo, hi, 0, 1, 2, 3, 4, 5, 6, 7,
                                               8, 9, 10, 11, 12, 13, 14, 15);
      afrag[t][kt] = __builtin_bit_cast(v16bf, v);
    }
  }

  float rs[2][8];
  #pragma unroll
  for (int t = 0; t < 2; ++t)
    #pragma unroll
    for (int r = 0; r < 8; ++r) rs[t][r] = 0.0f;

  for (int ct = 0; ct < 8; ++ct) {
    // bf16 B 32x16 layout: lane = column, elems 0..7 -> K = 8*half..+7 ; elems 8..15 -> K = 16+8*half..+7
    const us8* bcol = (const us8*)(nB + (size_t)(colBase + ct * 16 + nn) * PDIM);
    v8f c0 = {}; v8f c1 = {};
    #pragma unroll
    for (int kt = 0; kt < 8; ++kt) {
      us8 lo = bcol[kt * 4 + half];
      us8 hi = bcol[kt * 4 + 2 + half];
      us16 v = __builtin_shufflevector(lo, hi, 0, 1, 2, 3, 4, 5, 6, 7,
                                               8, 9, 10, 11, 12, 13, 14, 15);
      v16bf bfrag = __builtin_bit_cast(v16bf, v);
      c0 = __builtin_amdgcn_wmma_f32_16x16x32_bf16(false, afrag[0][kt], false, bfrag,
                                                   (short)0, c0, false, false);
      c1 = __builtin_amdgcn_wmma_f32_16x16x32_bf16(false, afrag[1][kt], false, bfrag,
                                                   (short)0, c1, false, false);
    }
    float cpart = 0.0f;
    #pragma unroll
    for (int r = 0; r < 8; ++r) {
      float e0 = __expf(c0[r] * ITAU);
      float e1 = __expf(c1[r] * ITAU);
      rs[0][r] += e0;
      rs[1][r] += e1;
      cpart += e0 + e1;
    }
    if (wantCol) {
      cpart += __shfl_xor(cpart, 16, 32);
      if (half == 0) atomicAdd(&col_lds[ct * 16 + nn], cpart);
    }
  }

  #pragma unroll
  for (int t = 0; t < 2; ++t) {
    #pragma unroll
    for (int r = 0; r < 8; ++r) {
      float s = rs[t][r];
      s += __shfl_xor(s, 1, 32);
      s += __shfl_xor(s, 2, 32);
      s += __shfl_xor(s, 4, 32);
      s += __shfl_xor(s, 8, 32);
      if (nn == 0) atomicAdd(&rowsum[rowBase + t * 16 + r + 8 * half], s);
    }
  }

  if (wantCol) {
    __syncthreads();
    if (tid < 128) atomicAdd(&colsum[colBase + tid], col_lds[tid]);
  }
}

// ---------------------------------------------------------------------------
// Kernel 3: per-row loss and mean.
//   denom1_i = s11_i + s12row_i - exp(1/tau)   (diag of refl is exactly e^2)
//   denom2_i = s22_i + s12col_i - exp(1/tau)
//   loss_i   = 0.5*(log d1 + log d2) - (n1_i . n2_i)/tau
// One wave per row; scalar output accumulated with atomics.
// ---------------------------------------------------------------------------
__global__ __launch_bounds__(256)
void finalize_kernel(const unsigned short* __restrict__ n1,
                     const unsigned short* __restrict__ n2,
                     const float* __restrict__ s11, const float* __restrict__ s22,
                     const float* __restrict__ s12r, const float* __restrict__ s12c,
                     float* __restrict__ out)
{
  const int tid  = threadIdx.x;
  const int wid  = tid >> 5;
  const int lane = tid & 31;
  const int i = blockIdx.x * 8 + wid;

  const unsigned short* a = n1 + (size_t)i * PDIM;
  const unsigned short* b = n2 + (size_t)i * PDIM;
  float dot = 0.0f;
  #pragma unroll
  for (int t = 0; t < 8; ++t) {
    int k = lane + t * 32;
    dot += bf2f(a[k]) * bf2f(b[k]);
  }
  dot += __shfl_xor(dot, 1, 32);
  dot += __shfl_xor(dot, 2, 32);
  dot += __shfl_xor(dot, 4, 32);
  dot += __shfl_xor(dot, 8, 32);
  dot += __shfl_xor(dot, 16, 32);

  if (lane == 0) {
    float d1 = s11[i] + s12r[i] - EXP_INV_TAU;
    float d2 = s22[i] + s12c[i] - EXP_INV_TAU;
    float loss = 0.5f * (logf(d1) + logf(d2)) - dot * ITAU;
    atomicAdd(out, loss * (1.0f / (float)NROWS));
  }
}

// ---------------------------------------------------------------------------
extern "C" void kernel_launch(void* const* d_in, const int* in_sizes, int n_in,
                              void* d_out, int out_size, void* d_ws, size_t ws_size,
                              hipStream_t stream) {
  (void)in_sizes; (void)n_in; (void)out_size; (void)ws_size;
  const float* z1 = (const float*)d_in[0];
  const float* z2 = (const float*)d_in[1];
  const float* w1 = (const float*)d_in[2];
  const float* b1 = (const float*)d_in[3];
  const float* w2 = (const float*)d_in[4];
  const float* b2 = (const float*)d_in[5];

  char* ws = (char*)d_ws;
  unsigned short* n1 = (unsigned short*)ws;                                  // 8 MB
  unsigned short* n2 = (unsigned short*)(ws + (size_t)8 * 1024 * 1024);      // 8 MB
  float* s11  = (float*)(ws + (size_t)16 * 1024 * 1024);                     // 4 x 64 KB
  float* s22  = s11 + NROWS;
  float* s12r = s22 + NROWS;
  float* s12c = s12r + NROWS;

  hipMemsetAsync(s11, 0, (size_t)4 * NROWS * sizeof(float), stream);
  hipMemsetAsync(d_out, 0, sizeof(float), stream);

  proj_norm_kernel<<<dim3(NROWS / 16, 2), 256, 0, stream>>>(
      z1, z2, w1, b1, w2, b2, n1, n2);

  dim3 gg(NROWS / 128, NROWS / 256);
  gram_expsum_kernel<<<gg, 256, 0, stream>>>(n1, n1, s11, s11, 0);
  gram_expsum_kernel<<<gg, 256, 0, stream>>>(n2, n2, s22, s22, 0);
  gram_expsum_kernel<<<gg, 256, 0, stream>>>(n1, n2, s12r, s12c, 1);

  finalize_kernel<<<NROWS / 8, 256, 0, stream>>>(
      n1, n2, s11, s22, s12r, s12c, (float*)d_out);
}
